// InitLayer_14190571946518
// MI455X (gfx1250) — matrix operand order repaired and checked by reading, not compiled, not run on hardware
//
#include <hip/hip_runtime.h>
#include <hip/hip_bf16.h>
#include <hip/hip_fp16.h>

typedef __attribute__((ext_vector_type(16))) _Float16 v16h;
typedef __attribute__((ext_vector_type(8)))  _Float16 v8h;
typedef __attribute__((ext_vector_type(8)))  float    v8f;

#define E_EDGES 320000
#define N_NODES 10000
#define RMAXF   5.0f
#define OUTC    288

// f16 fragment-weight offsets in d_ws (units: halves)
#define OFF1 0        // W1: 5 ktiles x 8 ntiles x 512 = 20480 halves
#define OFF2 20480    // W2: 4 x 8 x 512 = 16384
#define OFF3 36864    // W3: 4 x 8 x 512 = 16384
#define OFF4 53248    // W_env: 4 x 6 x 512 = 12288  (total 131072 B)

// ---------------------------------------------------------------------------
// Pack a (Kreal x Ndim) row-major f32 weight into WMMA B-fragment layout f16,
// scale folded in, K zero-padded to Ktiles*32.
// B layout (16x16x32 f16): lane L -> n = L&15 ; element e<8 -> k = (L>>4)*8 + e ; e>=8 -> k = 16 + (L>>4)*8 + (e-8)
// Fragment-major: [(ktile*Ntiles + ntile)*32 + lane] -> v16h
// ---------------------------------------------------------------------------
__global__ void prep_weights(const float* __restrict__ W, _Float16* __restrict__ dst,
                             int Kreal, int Ktiles, int Ntiles, int Ndim, float scale) {
  int idx = blockIdx.x * blockDim.x + threadIdx.x;   // one thread per (frag, lane)
  int total = Ktiles * Ntiles * 32;
  if (idx >= total) return;
  int lane = idx & 31;
  int frag = idx >> 5;
  int n_t  = frag % Ntiles;
  int k_t  = frag / Ntiles;
  int n    = n_t * 16 + (lane & 15);
  int kh   = (lane >> 4) * 8;
  v16h out;
  #pragma unroll
  for (int e = 0; e < 16; ++e) {
    int k = k_t * 32 + kh + e + (e >= 8 ? 8 : 0);
    float v = (k < Kreal) ? W[k * Ndim + n] * scale : 0.0f;
    out[e] = (_Float16)v;
  }
  ((v16h*)dst)[idx] = out;
}

__global__ void zero_f32(float* __restrict__ p, long n) {
  long i = (long)blockIdx.x * blockDim.x + threadIdx.x;
  if (i < n) p[i] = 0.0f;
}

// ---------------------------------------------------------------------------
// Per-layer WMMA chain helper: acc[n] = sum_t A[t] x B[t,n]  (f32 accum)
// ---------------------------------------------------------------------------
template <int KT, int NT>
__device__ inline void layer_wmma(const v16h* A, const _Float16* wf, int lane, v8f* acc) {
  const v16h* B = (const v16h*)wf;
  #pragma unroll
  for (int n = 0; n < NT; ++n) {
    v8f c = {0.f, 0.f, 0.f, 0.f, 0.f, 0.f, 0.f, 0.f};
    #pragma unroll
    for (int t = 0; t < KT; ++t) {
      v16h b = B[(t * NT + n) * 32 + lane];
      c = __builtin_amdgcn_wmma_f32_16x16x32_f16(false, A[t], false, b,
                                                 (short)0, c, false, false);
    }
    acc[n] = c;
  }
}

// A layout (16x32 f16, M x K): lane L -> m = L&15, hh = L>>4 ;
//   e in 0..7  -> k = hh*8 + e ; e in 8..15 -> k = 16 + hh*8 + (e-8)
template <int KT>
__device__ inline void load_A_from_lds(const _Float16* st, int ST, int mrow, int hh, v16h* A) {
  #pragma unroll
  for (int t = 0; t < KT; ++t) {
    v8h lo = *(const v8h*)&st[mrow * ST + t * 32 + hh * 8];        // 16B aligned
    v8h hi = *(const v8h*)&st[mrow * ST + t * 32 + 16 + hh * 8];
    v16h a;
    #pragma unroll
    for (int e = 0; e < 8; ++e) { a[e] = lo[e]; a[8 + e] = hi[e]; }
    A[t] = a;
  }
}

// ---------------------------------------------------------------------------
// Fused per-edge pipeline. 8 waves/block, 1 wave = 16 edges, 2500 blocks.
// ---------------------------------------------------------------------------
__global__ __launch_bounds__(256) void fused_edges(
    const int*   __restrict__ edge_index,
    const float* __restrict__ edge_sh,
    const float* __restrict__ edge_length,
    const float* __restrict__ one_hot,
    const float* __restrict__ bessel_w,
    const _Float16* __restrict__ wf,
    float* __restrict__ out_lat,
    float* __restrict__ out_nodes) {
  // per-wave private LDS slice: max(16*136 f16, 16*100 f32) = 6400 B
  __shared__ __align__(16) unsigned char lds_raw[8 * 6400];

  const int tid  = threadIdx.x;
  const int wave = tid >> 5;
  const int lane = tid & 31;
  const int hh   = lane >> 4;    // which half of the wave
  const int mrow = lane & 15;
  const long base = ((long)blockIdx.x * 8 + wave) * 16;   // first edge of tile

  _Float16* st  = (_Float16*)(lds_raw + wave * 6400);  // f16 staging, stride 136
  float*    stw = (float*)   (lds_raw + wave * 6400);  // f32 env-w staging, stride 100
  const int ST = 136;

  // ---- polynomial cutoff for the 8 rows this lane owns in C/D fragments
  float cut[8];
  {
    const float* rl = edge_length + base + hh * 8;
    #pragma unroll
    for (int i = 0; i < 8; ++i) {
      float x  = rl[i] * (1.0f / RMAXF);
      float x2 = x * x, x3 = x2 * x, x6 = x3 * x3;
      float o  = 1.0f - 28.0f * x6 + 48.0f * x6 * x - 21.0f * x6 * x2;
      cut[i] = (x < 1.0f) ? o : 0.0f;
    }
  }

  // ---- build layer-1 A fragments: one-hot (K 0..127) + bessel (K 128..135), pad->160
  v16h A1[5];
  {
    const float* rowp = one_hot + (base + mrow) * 128;
    #pragma unroll
    for (int t = 0; t < 4; ++t) {
      const float4* p0 = (const float4*)(rowp + t * 32 + hh * 8);
      const float4* p1 = (const float4*)(rowp + t * 32 + 16 + hh * 8);
      float4 a0 = p0[0], a1 = p0[1], b0 = p1[0], b1 = p1[1];
      v16h a;
      a[0]  = (_Float16)a0.x; a[1]  = (_Float16)a0.y; a[2]  = (_Float16)a0.z; a[3]  = (_Float16)a0.w;
      a[4]  = (_Float16)a1.x; a[5]  = (_Float16)a1.y; a[6]  = (_Float16)a1.z; a[7]  = (_Float16)a1.w;
      a[8]  = (_Float16)b0.x; a[9]  = (_Float16)b0.y; a[10] = (_Float16)b0.z; a[11] = (_Float16)b0.w;
      a[12] = (_Float16)b1.x; a[13] = (_Float16)b1.y; a[14] = (_Float16)b1.z; a[15] = (_Float16)b1.w;
      A1[t] = a;
    }
    float r   = edge_length[base + mrow];
    float s   = (2.0f / RMAXF) / r;
    float msk = (hh == 0) ? 1.0f : 0.0f;    // half 1 covers K 136..143 -> zero pad
    v16h a;
    #pragma unroll
    for (int e = 0; e < 16; ++e) a[e] = (_Float16)0.0f;
    #pragma unroll
    for (int j = 0; j < 8; ++j)
      a[j] = (_Float16)(msk * s * __sinf(bessel_w[j] * (r * (1.0f / RMAXF))));
    A1[4] = a;
  }

  v8f acc[8];

  // ---- layer 1: (16x160) x (160x128), SiLU -> LDS f16
  layer_wmma<5, 8>(A1, wf + OFF1, lane, acc);
  #pragma unroll
  for (int n = 0; n < 8; ++n)
    #pragma unroll
    for (int i = 0; i < 8; ++i) {
      float x = acc[n][i];
      st[(i + 8 * hh) * ST + n * 16 + mrow] = (_Float16)(x / (1.0f + __expf(-x)));
    }
  __syncthreads();

  // ---- layer 2
  v16h A2[4];
  load_A_from_lds<4>(st, ST, mrow, hh, A2);
  __syncthreads();
  layer_wmma<4, 8>(A2, wf + OFF2, lane, acc);
  #pragma unroll
  for (int n = 0; n < 8; ++n)
    #pragma unroll
    for (int i = 0; i < 8; ++i) {
      float x = acc[n][i];
      st[(i + 8 * hh) * ST + n * 16 + mrow] = (_Float16)(x / (1.0f + __expf(-x)));
    }
  __syncthreads();

  // ---- layer 3 -> latents = cutoff * lat ; store to d_out and re-stage as f16
  v16h A3[4];
  load_A_from_lds<4>(st, ST, mrow, hh, A3);
  __syncthreads();
  layer_wmma<4, 8>(A3, wf + OFF3, lane, acc);
  #pragma unroll
  for (int n = 0; n < 8; ++n)
    #pragma unroll
    for (int i = 0; i < 8; ++i) {
      float v = acc[n][i] * cut[i];
      out_lat[(size_t)(base + i + 8 * hh) * 128 + n * 16 + mrow] = v;
      st[(i + 8 * hh) * ST + n * 16 + mrow] = (_Float16)v;
    }
  __syncthreads();

  // ---- layer 4 (env weights): (16x128) x (128x96) -> LDS f32
  v16h A4[4];
  load_A_from_lds<4>(st, ST, mrow, hh, A4);
  __syncthreads();
  layer_wmma<4, 6>(A4, wf + OFF4, lane, acc);
  #pragma unroll
  for (int n = 0; n < 6; ++n)
    #pragma unroll
    for (int i = 0; i < 8; ++i)
      stw[(i + 8 * hh) * 100 + n * 16 + mrow] = acc[n][i];
  __syncthreads();

  // ---- SH outer product + scatter-add into node features
  int dsti[16];
  #pragma unroll
  for (int m = 0; m < 16; ++m) dsti[m] = edge_index[base + m];  // row 0 of edge_index

  const float inv_avg = 0.17677669529663687f;  // 32^-0.5
  for (int c32 = 0; c32 < 9; ++c32) {
    int col = c32 * 32 + lane;                 // 0..287
    int path, u, shidx;
    if (col < 32)       { path = 0; u = col;             shidx = 0; }
    else if (col < 128) { int t2 = col - 32;  path = 1; u = t2 / 3; shidx = 1 + t2 % 3; }
    else                { int t2 = col - 128; path = 2; u = t2 / 5; shidx = 4 + t2 % 5; }
    int wcol = path * 32 + u;
    #pragma unroll
    for (int m = 0; m < 16; ++m) {
      float val = stw[m * 100 + wcol] * edge_sh[(base + m) * 9 + shidx] * inv_avg;
      unsafeAtomicAdd(&out_nodes[(size_t)dsti[m] * OUTC + col], val);
    }
  }
}

// ---------------------------------------------------------------------------
extern "C" void kernel_launch(void* const* d_in, const int* in_sizes, int n_in,
                              void* d_out, int out_size, void* d_ws, size_t ws_size,
                              hipStream_t stream) {
  const int*   edge_index  = (const int*)  d_in[0];
  // d_in[1] atom_type, d_in[2] bond_type: unused by the reference
  const float* edge_sh     = (const float*)d_in[3];
  const float* edge_length = (const float*)d_in[4];
  const float* one_hot     = (const float*)d_in[5];
  const float* bessel_w    = (const float*)d_in[6];
  const float* W1          = (const float*)d_in[7];
  const float* W2          = (const float*)d_in[8];
  const float* W3          = (const float*)d_in[9];
  const float* W4          = (const float*)d_in[10];

  float* out_lat   = (float*)d_out;
  float* out_nodes = out_lat + (size_t)E_EDGES * 128;
  _Float16* wf     = (_Float16*)d_ws;

  const float s1 = 0.08574929257125442f;   // 136^-0.5
  const float s2 = 0.08838834764831843f;   // 128^-0.5

  prep_weights<<<(5 * 8 * 32 + 255) / 256, 256, 0, stream>>>(W1, wf + OFF1, 136, 5, 8, 128, s1);
  prep_weights<<<(4 * 8 * 32 + 255) / 256, 256, 0, stream>>>(W2, wf + OFF2, 128, 4, 8, 128, s2);
  prep_weights<<<(4 * 8 * 32 + 255) / 256, 256, 0, stream>>>(W3, wf + OFF3, 128, 4, 8, 128, s2);
  prep_weights<<<(4 * 6 * 32 + 255) / 256, 256, 0, stream>>>(W4, wf + OFF4, 128, 4, 6, 96,  s2);

  long nz = (long)N_NODES * OUTC;
  zero_f32<<<(nz + 255) / 256, 256, 0, stream>>>(out_nodes, nz);

  // 320000 edges / 16 per wave / 8 waves per block = 2500 blocks exactly
  fused_edges<<<2500, 256, 0, stream>>>(edge_index, edge_sh, edge_length, one_hot,
                                        bessel_w, wf, out_lat, out_nodes);
}
